// LanguageModel_86629490360952
// MI455X (gfx1250) — compile-verified, matrix operand order checked by reading, not run on hardware
//
#include <hip/hip_runtime.h>
#include <hip/hip_bf16.h>
#include <math.h>

// ---------------- model dimensions ----------------
#define BB 8
#define SS 1024
#define LL 12
#define DD 768
#define HH 12
#define FF 3072
#define VV 21128
#define DH 64
#define TITLE 64
#define CONT 960
#define NTOK (BB * CONT)        // 7680
#define VCHUNK 2641             // V = 8 * 2641
#define LDLOG 2656              // padded logits row stride

typedef __attribute__((ext_vector_type(16))) _Float16     v16h;
typedef __attribute__((ext_vector_type(8)))  float        v8f;
typedef __attribute__((ext_vector_type(4)))  unsigned int v4u;
typedef __attribute__((ext_vector_type(8)))  int          v8i;
typedef __attribute__((ext_vector_type(4)))  int          v4i;

// gfx1250 WMMA 16x16x32 f16 fragment K-index map for element e of a v16h,
// for half-wave hi (0: lanes 0-15, 1: lanes 16-31).
#define KMAP(e, hi) (((((e) >> 1) < 4) ? 0 : 16) + (hi) * 8 + ((((e) >> 1) & 3) * 2) + ((e) & 1))

#define WMMA_F16(a, b, c) \
  __builtin_amdgcn_wmma_f32_16x16x32_f16(false, (a), false, (b), (short)0, (c), false, false)

// =====================================================================
// TDM: load a 2D fp16 tile (nrows x 32 halves) from global to LDS.
// LDS rows padded by 1 DWORD -> effective LDS row stride 34 halves
// (pad_interval = 16 DWORDs = one 32-half row, pad_amount = 1 DWORD).
// tensor_dim1 = rows_avail gives hardware zero-fill past the edge.
// D# packing per CDNA5 ISA 8.3-8.6 (2D, groups 2/3 zero).
// This toolchain's builtin takes 6 args (extra i32x8 group, zero-filled).
// =====================================================================
static __device__ __forceinline__ void tdm_load_tile(
    unsigned lds_off, const _Float16* gptr, unsigned rows_avail,
    unsigned long stride_h) {
  const unsigned long ga = (unsigned long)(const void*)gptr;
  const unsigned dim0 = 32u;       // tensor_dim0 (halves)
  const unsigned dim1 = rows_avail;
  const unsigned tile0 = 32u, tile1 = 64u;

  v4u g0;
  g0[0] = 1u;                                            // count=1
  g0[1] = lds_off;                                       // lds_addr (bytes)
  g0[2] = (unsigned)(ga & 0xFFFFFFFFu);                  // global_addr lo
  g0[3] = (unsigned)((ga >> 32) & 0x1FFFFFFu) | (2u << 30);  // addr hi | type=2

  v8i g1;
  // data_size=1(2B) | pad_enable | pad_interval=3(16 DW) | pad_amount=0(1 DW)
  g1[0] = (int)((1u << 16) | (1u << 20) | (3u << 22));
  g1[1] = (int)((dim0 & 0xFFFFu) << 16);
  g1[2] = (int)(((dim0 >> 16) & 0xFFFFu) | ((dim1 & 0xFFFFu) << 16));
  g1[3] = (int)(((dim1 >> 16) & 0xFFFFu) | ((tile0 & 0xFFFFu) << 16));
  g1[4] = (int)(tile1 & 0xFFFFu);                        // tile_dim2 = 0
  g1[5] = (int)(stride_h & 0xFFFFFFFFu);                 // dim0_stride lo
  g1[6] = (int)((stride_h >> 32) & 0xFFFFu);             // dim0_stride hi
  g1[7] = 0;

  v4i g2; g2[0] = 0; g2[1] = 0; g2[2] = 0; g2[3] = 0;
  v4i g3; g3[0] = 0; g3[1] = 0; g3[2] = 0; g3[3] = 0;
  v8i g4;
#pragma unroll
  for (int i = 0; i < 8; ++i) g4[i] = 0;
  __builtin_amdgcn_tensor_load_to_lds(g0, g1, g2, g3, g4, 0);
}

// =====================================================================
// Weight convert + transpose: W fp32 [K,N] -> Wt fp16 [N,K].
// 32x32 tiles through LDS for coalescing. K % 32 == 0; N guarded.
// =====================================================================
__global__ __launch_bounds__(256) void convT_kernel(
    const float* __restrict__ W, _Float16* __restrict__ Wt, int K, int N) {
  __shared__ float tile[32][33];
  const int tx = threadIdx.x & 31, ty = threadIdx.x >> 5;
  const int k0 = blockIdx.y * 32, n0 = blockIdx.x * 32;
#pragma unroll
  for (int j = 0; j < 4; ++j) {
    const int k = k0 + ty + j * 8, n = n0 + tx;
    tile[ty + j * 8][tx] = (n < N) ? W[(long)k * N + n] : 0.f;
  }
  __syncthreads();
#pragma unroll
  for (int j = 0; j < 4; ++j) {
    const int n = n0 + ty + j * 8, k = k0 + tx;
    if (n < N) Wt[(long)n * K + k] = (_Float16)tile[tx][ty + j * 8];
  }
}

// =====================================================================
// Embedding + LayerNorm -> fp16 activations
// =====================================================================
__global__ __launch_bounds__(256) void embed_ln_kernel(
    const int* __restrict__ x, const float* __restrict__ wemb,
    const float* __restrict__ pemb, const float* __restrict__ temb,
    const float* __restrict__ g, const float* __restrict__ bta,
    _Float16* __restrict__ out) {
  const int tok = blockIdx.x;
  const int s = tok % SS;
  const long wid = x[tok];
  __shared__ float red[256];
  float vals[3];
  float sum = 0.f;
#pragma unroll
  for (int j = 0; j < 3; ++j) {
    const int e = threadIdx.x + j * 256;
    const float v = wemb[wid * DD + e] + pemb[(long)s * DD + e] + temb[e];
    vals[j] = v;
    sum += v;
  }
  red[threadIdx.x] = sum;
  __syncthreads();
  for (int st = 128; st > 0; st >>= 1) {
    if (threadIdx.x < st) red[threadIdx.x] += red[threadIdx.x + st];
    __syncthreads();
  }
  const float mean = red[0] / (float)DD;
  __syncthreads();
  float vs = 0.f;
#pragma unroll
  for (int j = 0; j < 3; ++j) { const float d = vals[j] - mean; vs += d * d; }
  red[threadIdx.x] = vs;
  __syncthreads();
  for (int st = 128; st > 0; st >>= 1) {
    if (threadIdx.x < st) red[threadIdx.x] += red[threadIdx.x + st];
    __syncthreads();
  }
  const float rstd = rsqrtf(red[0] / (float)DD + 1e-12f);
#pragma unroll
  for (int j = 0; j < 3; ++j) {
    const int e = threadIdx.x + j * 256;
    out[(long)tok * DD + e] = (_Float16)((vals[j] - mean) * rstd * g[e] + bta[e]);
  }
}

// =====================================================================
// Residual add + LayerNorm (fp16 in/out, fp32 math)
// =====================================================================
__global__ __launch_bounds__(256) void add_ln_kernel(
    const _Float16* __restrict__ a, const _Float16* __restrict__ r,
    const float* __restrict__ g, const float* __restrict__ bta,
    _Float16* __restrict__ out) {
  const long tok = blockIdx.x;
  __shared__ float red[256];
  float vals[3];
  float sum = 0.f;
#pragma unroll
  for (int j = 0; j < 3; ++j) {
    const int e = threadIdx.x + j * 256;
    const float v = (float)a[tok * DD + e] + (float)r[tok * DD + e];
    vals[j] = v;
    sum += v;
  }
  red[threadIdx.x] = sum;
  __syncthreads();
  for (int st = 128; st > 0; st >>= 1) {
    if (threadIdx.x < st) red[threadIdx.x] += red[threadIdx.x + st];
    __syncthreads();
  }
  const float mean = red[0] / (float)DD;
  __syncthreads();
  float vs = 0.f;
#pragma unroll
  for (int j = 0; j < 3; ++j) { const float d = vals[j] - mean; vs += d * d; }
  red[threadIdx.x] = vs;
  __syncthreads();
  for (int st = 128; st > 0; st >>= 1) {
    if (threadIdx.x < st) red[threadIdx.x] += red[threadIdx.x + st];
    __syncthreads();
  }
  const float rstd = rsqrtf(red[0] / (float)DD + 1e-12f);
#pragma unroll
  for (int j = 0; j < 3; ++j) {
    const int e = threadIdx.x + j * 256;
    out[tok * DD + e] = (_Float16)((vals[j] - mean) * rstd * g[e] + bta[e]);
  }
}

// =====================================================================
// TDM + WMMA GEMM:  C[M,N] = act( A[M,K] @ Wt[N,K]^T + bias[N] ), fp16 I/O.
// 256 threads = 8 waves; 64x64 output tile; K steps of 32.
// Wave 0 drives the Tensor Data Mover (double-buffered LDS tiles with
// hardware padding -> 34-half row stride, conflict-free fragment reads);
// all 8 waves do two WMMAs per step.
// Requires: M % 64 == 0, K % 32 == 0. N ragged edge zero-filled by TDM.
// =====================================================================
__global__ __launch_bounds__(256) void gemm_kernel(
    const _Float16* __restrict__ A, int lda,
    const _Float16* __restrict__ Wt,   // [N,K] row-major (pre-transposed)
    const float* __restrict__ bias,
    _Float16* __restrict__ C, int ldc,
    int N, int K, int act) {
  __shared__ _Float16 As[2][64 * 34];
  __shared__ _Float16 Ws[2][64 * 34];

  const int tid = threadIdx.x;
  const int lane = tid & 31, wave = tid >> 5;
  const int lo = lane & 15, hi = lane >> 4;
  const int cw = wave & 3, pw = wave >> 2;
  const long rbase = (long)blockIdx.y * 64;
  const int cbase = blockIdx.x * 64;
  const unsigned rowsWT = (unsigned)(N - cbase);  // tensor_dim1 (free zero-fill)

  v8f acc0, acc1;
#pragma unroll
  for (int e = 0; e < 8; ++e) { acc0[e] = 0.f; acc1[e] = 0.f; }

  const int nk = K / 32;
  if (wave == 0) {
    tdm_load_tile((unsigned)(unsigned long)(void*)&As[0][0],
                  A + rbase * lda, 64u, (unsigned long)lda);
    tdm_load_tile((unsigned)(unsigned long)(void*)&Ws[0][0],
                  Wt + (long)cbase * K, rowsWT, (unsigned long)K);
  }

  for (int t = 0; t < nk; ++t) {
    const int cur = t & 1;
    if (wave == 0) {
      if (t + 1 < nk) {
        const int k1 = (t + 1) * 32;
        tdm_load_tile((unsigned)(unsigned long)(void*)&As[cur ^ 1][0],
                      A + rbase * lda + k1, 64u, (unsigned long)lda);
        tdm_load_tile((unsigned)(unsigned long)(void*)&Ws[cur ^ 1][0],
                      Wt + (long)cbase * K + k1, rowsWT, (unsigned long)K);
        __builtin_amdgcn_s_wait_tensorcnt(2);
      } else {
        __builtin_amdgcn_s_wait_tensorcnt(0);
      }
    }
    __syncthreads();

    v16h fb, fa0, fa1;
#pragma unroll
    for (int e = 0; e < 16; ++e) {
      const int kk = KMAP(e, hi);
      fb[e]  = Ws[cur][(cw * 16 + lo) * 34 + kk];
      fa0[e] = As[cur][(pw * 32 + lo) * 34 + kk];
      fa1[e] = As[cur][(pw * 32 + 16 + lo) * 34 + kk];
    }
    acc0 = WMMA_F16(fa0, fb, acc0);
    acc1 = WMMA_F16(fa1, fb, acc1);
    __syncthreads();
  }

  const int col = cbase + cw * 16 + lo;
  if (col < N) {
    const float bval = bias[col];
#pragma unroll
    for (int r = 0; r < 8; ++r) {
      float x0 = acc0[r] + bval;
      float x1 = acc1[r] + bval;
      if (act) {
        x0 = 0.5f * x0 * (1.f + erff(x0 * 0.70710678118f));
        x1 = 0.5f * x1 * (1.f + erff(x1 * 0.70710678118f));
      }
      C[(rbase + pw * 32 + r + 8 * hi) * (long)ldc + col] = (_Float16)x0;
      C[(rbase + pw * 32 + 16 + r + 8 * hi) * (long)ldc + col] = (_Float16)x1;
    }
  }
}

// =====================================================================
// Fused flash attention (prefix-LM mask), one wave per (b, h, 16-query tile).
// q,k,v,ctx: [B,S,D] fp16 with head offset h*64. fp32 online-softmax state.
// =====================================================================
__global__ __launch_bounds__(32) void attn_kernel(
    const _Float16* __restrict__ q, const _Float16* __restrict__ k,
    const _Float16* __restrict__ v, _Float16* __restrict__ ctx) {
  const int lane = threadIdx.x;
  const int lo = lane & 15, hi = lane >> 4;
  const int qt = blockIdx.x, h = blockIdx.y, b = blockIdx.z;
  const int q0 = qt * 16;
  const long baseQ = ((long)b * SS + q0) * DD + h * DH;

  __shared__ _Float16 pbuf[16 * 34];

  v16h aq0, aq1;
#pragma unroll
  for (int e = 0; e < 16; ++e) {
    const int kk = KMAP(e, hi);
    aq0[e] = q[baseQ + (long)lo * DD + kk];
    aq1[e] = q[baseQ + (long)lo * DD + 32 + kk];
  }

  float mrow[8], lrow[8];
  v8f o[4];
#pragma unroll
  for (int r = 0; r < 8; ++r) { mrow[r] = -3.0e38f; lrow[r] = 0.f; }
#pragma unroll
  for (int t = 0; t < 4; ++t)
#pragma unroll
    for (int e = 0; e < 8; ++e) o[t][e] = 0.f;

  for (int j0 = 0; j0 < SS; j0 += 32) {
    float sc[2][8];
#pragma unroll
    for (int t = 0; t < 2; ++t) {
      const long baseK = ((long)b * SS + j0 + t * 16 + lo) * DD + h * DH;
      v16h bk0, bk1;
#pragma unroll
      for (int e = 0; e < 16; ++e) {
        const int kk = KMAP(e, hi);
        bk0[e] = k[baseK + kk];
        bk1[e] = k[baseK + 32 + kk];
      }
      v8f sacc;
#pragma unroll
      for (int e = 0; e < 8; ++e) sacc[e] = 0.f;
      sacc = WMMA_F16(aq0, bk0, sacc);
      sacc = WMMA_F16(aq1, bk1, sacc);
#pragma unroll
      for (int r = 0; r < 8; ++r) {
        const int qi = q0 + r + 8 * hi;
        const int kj = j0 + t * 16 + lo;
        const float s = sacc[r] * 0.125f;  // 1/sqrt(64)
        const bool vis = (kj <= qi) || (qi < TITLE && kj < TITLE);
        sc[t][r] = vis ? s : (s - 1.0e9f);
      }
    }
    float ps[2][8];
#pragma unroll
    for (int r = 0; r < 8; ++r) {
      float pm = fmaxf(sc[0][r], sc[1][r]);
      pm = fmaxf(pm, __shfl_xor(pm, 1));
      pm = fmaxf(pm, __shfl_xor(pm, 2));
      pm = fmaxf(pm, __shfl_xor(pm, 4));
      pm = fmaxf(pm, __shfl_xor(pm, 8));
      const float newm = fmaxf(mrow[r], pm);
      const float alpha = __expf(mrow[r] - newm);
      const float p0 = __expf(sc[0][r] - newm);
      const float p1 = __expf(sc[1][r] - newm);
      float psum = p0 + p1;
      psum += __shfl_xor(psum, 1);
      psum += __shfl_xor(psum, 2);
      psum += __shfl_xor(psum, 4);
      psum += __shfl_xor(psum, 8);
      lrow[r] = lrow[r] * alpha + psum;
      mrow[r] = newm;
      ps[0][r] = p0;
      ps[1][r] = p1;
#pragma unroll
      for (int t = 0; t < 4; ++t) o[t][r] *= alpha;
    }
    __syncthreads();
#pragma unroll
    for (int t = 0; t < 2; ++t)
#pragma unroll
      for (int r = 0; r < 8; ++r)
        pbuf[(r + 8 * hi) * 34 + t * 16 + lo] = (_Float16)ps[t][r];
    __syncthreads();
    v16h pa;
#pragma unroll
    for (int e = 0; e < 16; ++e) pa[e] = pbuf[lo * 34 + KMAP(e, hi)];

#pragma unroll
    for (int nt = 0; nt < 4; ++nt) {
      v16h vfrag;
#pragma unroll
      for (int e = 0; e < 16; ++e) {
        const int kk = KMAP(e, hi);
        vfrag[e] = v[((long)b * SS + j0 + kk) * DD + h * DH + nt * 16 + lo];
      }
      o[nt] = WMMA_F16(pa, vfrag, o[nt]);
    }
  }
#pragma unroll
  for (int nt = 0; nt < 4; ++nt)
#pragma unroll
    for (int r = 0; r < 8; ++r)
      ctx[((long)b * SS + q0 + r + 8 * hi) * DD + h * DH + nt * 16 + lo] =
          (_Float16)(o[nt][r] / lrow[r]);
}

// =====================================================================
// Classifier-side helper kernels
// =====================================================================
__global__ __launch_bounds__(256) void gather_kernel(const _Float16* __restrict__ h,
                                                     _Float16* __restrict__ hcls) {
  const long i = (long)blockIdx.x * 256 + threadIdx.x;
  if (i >= (long)NTOK * DD) return;
  const long row = i / DD, col = i % DD;
  const long b = row / CONT, c = row % CONT;
  hcls[i] = h[((b * SS) + TITLE + c) * DD + col];
}

__global__ __launch_bounds__(256) void lse_init_kernel(float* m, float* sf, float* tg) {
  const int t = blockIdx.x * 256 + threadIdx.x;
  if (t < NTOK) { m[t] = -3.0e38f; sf[t] = 0.f; tg[t] = 0.f; }
}

__global__ __launch_bounds__(256) void lse_update_kernel(
    const _Float16* __restrict__ logits, int ldc, int ncols, int off,
    const int* __restrict__ y, float* m, float* sf, float* tg) {
  const int t = blockIdx.x * 256 + threadIdx.x;
  if (t >= NTOK) return;
  const _Float16* row = logits + (long)t * ldc;
  float cm = -3.0e38f;
  for (int j = 0; j < ncols; ++j) cm = fmaxf(cm, (float)row[j]);
  const float newm = fmaxf(m[t], cm);
  float ssum = sf[t] * __expf(m[t] - newm);
  for (int j = 0; j < ncols; ++j) ssum += __expf((float)row[j] - newm);
  m[t] = newm;
  sf[t] = ssum;
  const int loc = y[t] - off;
  if (loc >= 0 && loc < ncols) tg[t] = (float)row[loc];
}

__global__ __launch_bounds__(256) void nll_kernel(
    const int* __restrict__ y, const float* __restrict__ m,
    const float* __restrict__ sf, const float* __restrict__ tg,
    float* __restrict__ out) {
  __shared__ float rl[256];
  __shared__ float rc[256];
  float loss = 0.f, cnt = 0.f;
  for (int t = threadIdx.x; t < NTOK; t += 256) {
    if (y[t] != -100) {
      loss += m[t] + logf(sf[t]) - tg[t];
      cnt += 1.f;
    }
  }
  rl[threadIdx.x] = loss;
  rc[threadIdx.x] = cnt;
  __syncthreads();
  for (int st = 128; st > 0; st >>= 1) {
    if (threadIdx.x < st) {
      rl[threadIdx.x] += rl[threadIdx.x + st];
      rc[threadIdx.x] += rc[threadIdx.x + st];
    }
    __syncthreads();
  }
  if (threadIdx.x == 0) out[0] = rl[0] / fmaxf(rc[0], 1.f);
}

// =====================================================================
// Host orchestration
// =====================================================================
extern "C" void kernel_launch(void* const* d_in, const int* in_sizes, int n_in,
                              void* d_out, int out_size, void* d_ws, size_t ws_size,
                              hipStream_t stream) {
  (void)in_sizes; (void)n_in; (void)out_size; (void)ws_size;

  const int* x = (const int*)d_in[0];
  const int* y = (const int*)d_in[1];
  const float* wemb = (const float*)d_in[2];
  const float* pemb = (const float*)d_in[3];
  const float* temb = (const float*)d_in[4];
  const float* elng = (const float*)d_in[5];
  const float* elnb = (const float*)d_in[6];
  const float* Wq = (const float*)d_in[7];
  const float* bq = (const float*)d_in[8];
  const float* Wk = (const float*)d_in[9];
  const float* bk = (const float*)d_in[10];
  const float* Wv = (const float*)d_in[11];
  const float* bv = (const float*)d_in[12];
  const float* Wo = (const float*)d_in[13];
  const float* bo = (const float*)d_in[14];
  const float* ln1g = (const float*)d_in[15];
  const float* ln1b = (const float*)d_in[16];
  const float* W1 = (const float*)d_in[17];
  const float* b1 = (const float*)d_in[18];
  const float* W2 = (const float*)d_in[19];
  const float* b2 = (const float*)d_in[20];
  const float* ln2g = (const float*)d_in[21];
  const float* ln2b = (const float*)d_in[22];
  const float* clsW = (const float*)d_in[23];
  const float* clsb = (const float*)d_in[24];

  const int M = BB * SS;  // 8192

  // ---------------- workspace carve-out ----------------
  char* ws = (char*)d_ws;
  const size_t SZ_H = (size_t)M * DD * sizeof(_Float16);          // 12.6 MB
  const size_t SZ_F1 = (size_t)M * FF * sizeof(_Float16);         // 50.3 MB
  const size_t SZ_HC = (size_t)NTOK * DD * sizeof(_Float16);
  const size_t SZ_LG = (size_t)NTOK * LDLOG * sizeof(_Float16);
  const size_t SZ_WDD = (size_t)LL * DD * DD * sizeof(_Float16);  // 14.2 MB
  const size_t SZ_WDF = (size_t)LL * DD * FF * sizeof(_Float16);  // 56.6 MB
  const size_t SZ_WCL = (size_t)VV * DD * sizeof(_Float16);       // 32.5 MB
  size_t off = 0;
  _Float16* hA   = (_Float16*)(ws + off); off += SZ_H;
  _Float16* hB   = (_Float16*)(ws + off); off += SZ_H;
  _Float16* qb   = (_Float16*)(ws + off); off += SZ_H;
  _Float16* kb   = (_Float16*)(ws + off); off += SZ_H;
  _Float16* vb   = (_Float16*)(ws + off); off += SZ_H;
  _Float16* ctx  = (_Float16*)(ws + off); off += SZ_H;
  _Float16* tmp  = (_Float16*)(ws + off); off += SZ_H;
  _Float16* f1   = (_Float16*)(ws + off); off += SZ_F1;
  _Float16* hcls = (_Float16*)(ws + off); off += SZ_HC;
  _Float16* lgt  = (_Float16*)(ws + off); off += SZ_LG;
  _Float16* wqT  = (_Float16*)(ws + off); off += SZ_WDD;
  _Float16* wkT  = (_Float16*)(ws + off); off += SZ_WDD;
  _Float16* wvT  = (_Float16*)(ws + off); off += SZ_WDD;
  _Float16* woT  = (_Float16*)(ws + off); off += SZ_WDD;
  _Float16* w1T  = (_Float16*)(ws + off); off += SZ_WDF;
  _Float16* w2T  = (_Float16*)(ws + off); off += SZ_WDF;
  _Float16* wcT  = (_Float16*)(ws + off); off += SZ_WCL;
  float* mst = (float*)(ws + off); off += NTOK * sizeof(float);
  float* sst = (float*)(ws + off); off += NTOK * sizeof(float);
  float* tst = (float*)(ws + off); off += NTOK * sizeof(float);

  // ---------------- one-shot weight convert+transpose (fp32->fp16) --------
  const dim3 gDD(DD / 32, DD / 32);
  const dim3 gDF(FF / 32, DD / 32);   // W1 [D,F] -> [F,D]
  const dim3 gFD(DD / 32, FF / 32);   // W2 [F,D] -> [D,F]
  const dim3 gCV((VV + 31) / 32, DD / 32);
  for (int l = 0; l < LL; ++l) {
    convT_kernel<<<gDD, 256, 0, stream>>>(Wq + (long)l * DD * DD, wqT + (long)l * DD * DD, DD, DD);
    convT_kernel<<<gDD, 256, 0, stream>>>(Wk + (long)l * DD * DD, wkT + (long)l * DD * DD, DD, DD);
    convT_kernel<<<gDD, 256, 0, stream>>>(Wv + (long)l * DD * DD, wvT + (long)l * DD * DD, DD, DD);
    convT_kernel<<<gDD, 256, 0, stream>>>(Wo + (long)l * DD * DD, woT + (long)l * DD * DD, DD, DD);
    convT_kernel<<<gDF, 256, 0, stream>>>(W1 + (long)l * DD * FF, w1T + (long)l * DD * FF, DD, FF);
    convT_kernel<<<gFD, 256, 0, stream>>>(W2 + (long)l * FF * DD, w2T + (long)l * FF * DD, FF, DD);
  }
  convT_kernel<<<gCV, 256, 0, stream>>>(clsW, wcT, DD, VV);

  // ---------------- embedding ----------------
  embed_ln_kernel<<<M, 256, 0, stream>>>(x, wemb, pemb, temb, elng, elnb, hA);

  // ---------------- transformer layers ----------------
  const dim3 gProjD(DD / 64, M / 64);
  const dim3 gProjF(FF / 64, M / 64);
  const dim3 gAttn(SS / 16, HH, BB);

  for (int l = 0; l < LL; ++l) {
    const long oDD = (long)l * DD * DD;
    const long oDF = (long)l * DD * FF;

    gemm_kernel<<<gProjD, 256, 0, stream>>>(hA, DD, wqT + oDD, bq + l * DD, qb, DD, DD, DD, 0);
    gemm_kernel<<<gProjD, 256, 0, stream>>>(hA, DD, wkT + oDD, bk + l * DD, kb, DD, DD, DD, 0);
    gemm_kernel<<<gProjD, 256, 0, stream>>>(hA, DD, wvT + oDD, bv + l * DD, vb, DD, DD, DD, 0);

    attn_kernel<<<gAttn, 32, 0, stream>>>(qb, kb, vb, ctx);

    gemm_kernel<<<gProjD, 256, 0, stream>>>(ctx, DD, woT + oDD, bo + l * DD, tmp, DD, DD, DD, 0);
    add_ln_kernel<<<M, 256, 0, stream>>>(hA, tmp, ln1g + l * DD, ln1b + l * DD, hB);

    gemm_kernel<<<gProjF, 256, 0, stream>>>(hB, DD, w1T + oDF, b1 + l * FF, f1, FF, FF, DD, 1);
    gemm_kernel<<<gProjD, 256, 0, stream>>>(f1, FF, w2T + oDF, b2 + l * DD, tmp, DD, DD, FF, 0);
    add_ln_kernel<<<M, 256, 0, stream>>>(hB, tmp, ln2g + l * DD, ln2b + l * DD, hA);
  }

  // ---------------- classifier (chunked over V) + NLL ----------------
  const long nGather = ((long)NTOK * DD + 255) / 256;
  gather_kernel<<<(int)nGather, 256, 0, stream>>>(hA, hcls);
  lse_init_kernel<<<(NTOK + 255) / 256, 256, 0, stream>>>(mst, sst, tst);

  const dim3 gCls((VCHUNK + 63) / 64, NTOK / 64);
  for (int c = 0; c < 8; ++c) {
    const int voff = c * VCHUNK;
    gemm_kernel<<<gCls, 256, 0, stream>>>(hcls, DD, wcT + (long)voff * DD, clsb + voff,
                                          lgt, LDLOG, VCHUNK, DD, 0);
    lse_update_kernel<<<(NTOK + 255) / 256, 256, 0, stream>>>(lgt, LDLOG, VCHUNK, voff,
                                                              y, mst, sst, tst);
  }
  nll_kernel<<<1, 256, 0, stream>>>(y, mst, sst, tst, (float*)d_out);
}
